// MogLSTM_70171175682755
// MI455X (gfx1250) — compile-verified, matrix operand order checked by reading, not run on hardware
//
#include <hip/hip_runtime.h>
#include <stdint.h>
#include <stddef.h>

// ---------------------------------------------------------------------------
// Mogrifier-LSTM (2 layers) persistent WMMA kernel for gfx1250 (MI455X).
// B=128, T=512, I=128, H=1024. bf16 WMMA (16x16x32) with f32 accumulate,
// fp32 recurrent state. Weights converted once to bf16 (~38MB, fits in L2).
// ---------------------------------------------------------------------------

#define Bdim 128
#define Tdim 512
#define Idim 128
#define Hdim 1024
#define NBLK 64
#define NTHR 256

typedef __bf16 bf16_t;
typedef __attribute__((ext_vector_type(16))) __bf16 bf16x16;
typedef __attribute__((ext_vector_type(8)))  __bf16 bf16x8;
typedef __attribute__((ext_vector_type(8)))  float  v8f;

static __device__ __forceinline__ bf16x16 load_ab16(const bf16_t* __restrict__ p, int k) {
  // lane-row pointer p already includes (half*8) K-offset; chunks at k and k+16
  bf16x8 lo = *reinterpret_cast<const bf16x8*>(p + k);
  bf16x8 hi = *reinterpret_cast<const bf16x8*>(p + k + 16);
  return __builtin_shufflevector(lo, hi, 0, 1, 2, 3, 4, 5, 6, 7,
                                         8, 9, 10, 11, 12, 13, 14, 15);
}

static __device__ __forceinline__ v8f wmma_bf16(bf16x16 a, bf16x16 b, v8f c) {
  return __builtin_amdgcn_wmma_f32_16x16x32_bf16(
      /*neg_a=*/false, a, /*neg_b=*/false, b,
      /*c_mod=*/(short)0, c, /*reuse_a=*/false, /*reuse_b=*/false);
}

// Fast activations: one v_exp_f32 + one v_rcp_f32 each (no IEEE div fixup).
static __device__ __forceinline__ float sigmoid_(float z) {
  return __builtin_amdgcn_rcpf(1.0f + __expf(-z));
}
static __device__ __forceinline__ float tanh_(float z) {
  // tanh(z) = 2*sigmoid(2z) - 1
  return __builtin_fmaf(2.0f, __builtin_amdgcn_rcpf(1.0f + __expf(-2.0f * z)), -1.0f);
}

// Grid-wide barrier: monotonic epoch counter in L2, release arrive / acquire spin.
static __device__ __forceinline__ void grid_barrier(unsigned* cnt, unsigned target) {
  __syncthreads();
  if (threadIdx.x == 0) {
    __hip_atomic_fetch_add(cnt, 1u, __ATOMIC_RELEASE, __HIP_MEMORY_SCOPE_AGENT);
    while (__hip_atomic_load(cnt, __ATOMIC_ACQUIRE, __HIP_MEMORY_SCOPE_AGENT) < target)
      __builtin_amdgcn_s_sleep(1);
  }
  __syncthreads();
}

// ---------------------------------------------------------------------------
// Mogrify phase: Out[m][n] = 2*sigmoid( sum_k A[m][k]*W[n][k] + bias[n] ) * Mult[m][n]
// A: (128 x K) bf16 row-major; W: (N x K) bf16 row-major (GEMM uses W^T).
// Writes fp32 Out and bf16 Outb (both 128 x N row-major).
// Force-inlined so loads trace to kernarg pointers -> global_load_* path.
// ---------------------------------------------------------------------------
static __device__ __forceinline__ void mogrify_phase(
    const bf16_t* __restrict__ A, int K,
    const bf16_t* __restrict__ W, const float* __restrict__ bias,
    const float* __restrict__ mult, int mult_stride,
    float* __restrict__ out, bf16_t* __restrict__ outb, int N,
    int gw, int nw)
{
  const int lane = threadIdx.x & 31;
  const int kh = lane >> 4;    // K half selector
  const int li = lane & 15;    // row (A) / column (W, C)
  const int ntiles = (Bdim / 16) * (N / 16);
  for (int t = gw; t < ntiles; t += nw) {
    const int m0 = (t & 7) << 4;
    const int n0 = (t >> 3) << 4;
    const bf16_t* arow = A + (size_t)(m0 + li) * K + kh * 8;
    const bf16_t* wrow = W + (size_t)(n0 + li) * K + kh * 8;
    v8f acc = {0.f, 0.f, 0.f, 0.f, 0.f, 0.f, 0.f, 0.f};
#pragma unroll 4
    for (int k = 0; k < K; k += 32)
      acc = wmma_bf16(load_ab16(arow, k), load_ab16(wrow, k), acc);
    const int nc = n0 + li;
    const float bz = bias[nc];
#pragma unroll
    for (int v = 0; v < 8; ++v) {
      const int mr = m0 + v + kh * 8;
      const size_t oi = (size_t)mr * N + nc;
      const float g = 2.0f * sigmoid_(acc[v] + bz);
      const float val = g * mult[(size_t)mr * mult_stride + nc];
      out[oi] = val;
      outb[oi] = (bf16_t)val;
    }
  }
}

// ---------------------------------------------------------------------------
// LSTM gate phase: gates = X@Wih^T + Hm@Whh^T + (bih+bhh); fused cell update.
// Each wave owns the same (m,n) tile of i,f,g,o (columns n, n+H, n+2H, n+3H),
// reusing the A tile across 4 WMMAs and updating c/h in-register.
// ---------------------------------------------------------------------------
static __device__ __forceinline__ void lstm_phase(
    const bf16_t* __restrict__ Xb, int K1,
    const bf16_t* __restrict__ Hb, int K2,
    const bf16_t* __restrict__ Wih, const bf16_t* __restrict__ Whh,
    const float* __restrict__ bih, const float* __restrict__ bhh,
    float* __restrict__ cst, float* __restrict__ hst, bf16_t* __restrict__ hstb,
    int gw, int nw)
{
  const int lane = threadIdx.x & 31;
  const int kh = lane >> 4;
  const int li = lane & 15;
  const int ntg = (Bdim / 16) * (Hdim / 16);  // 512 tile-groups
  for (int t = gw; t < ntg; t += nw) {
    const int m0 = (t & 7) << 4;
    const int n0 = (t >> 3) << 4;
    v8f a0 = {0.f,0.f,0.f,0.f,0.f,0.f,0.f,0.f};
    v8f a1 = a0, a2 = a0, a3 = a0;
    {
      const bf16_t* arow = Xb + (size_t)(m0 + li) * K1 + kh * 8;
      const bf16_t* w0 = Wih + (size_t)(n0 + 0 * Hdim + li) * K1 + kh * 8;
      const bf16_t* w1 = Wih + (size_t)(n0 + 1 * Hdim + li) * K1 + kh * 8;
      const bf16_t* w2 = Wih + (size_t)(n0 + 2 * Hdim + li) * K1 + kh * 8;
      const bf16_t* w3 = Wih + (size_t)(n0 + 3 * Hdim + li) * K1 + kh * 8;
#pragma unroll 2
      for (int k = 0; k < K1; k += 32) {
        bf16x16 av = load_ab16(arow, k);
        a0 = wmma_bf16(av, load_ab16(w0, k), a0);
        a1 = wmma_bf16(av, load_ab16(w1, k), a1);
        a2 = wmma_bf16(av, load_ab16(w2, k), a2);
        a3 = wmma_bf16(av, load_ab16(w3, k), a3);
      }
    }
    {
      const bf16_t* arow = Hb + (size_t)(m0 + li) * K2 + kh * 8;
      const bf16_t* w0 = Whh + (size_t)(n0 + 0 * Hdim + li) * K2 + kh * 8;
      const bf16_t* w1 = Whh + (size_t)(n0 + 1 * Hdim + li) * K2 + kh * 8;
      const bf16_t* w2 = Whh + (size_t)(n0 + 2 * Hdim + li) * K2 + kh * 8;
      const bf16_t* w3 = Whh + (size_t)(n0 + 3 * Hdim + li) * K2 + kh * 8;
#pragma unroll 2
      for (int k = 0; k < K2; k += 32) {
        bf16x16 av = load_ab16(arow, k);
        a0 = wmma_bf16(av, load_ab16(w0, k), a0);
        a1 = wmma_bf16(av, load_ab16(w1, k), a1);
        a2 = wmma_bf16(av, load_ab16(w2, k), a2);
        a3 = wmma_bf16(av, load_ab16(w3, k), a3);
      }
    }
    const int nc = n0 + li;
    const float bi = bih[nc + 0 * Hdim] + bhh[nc + 0 * Hdim];
    const float bf = bih[nc + 1 * Hdim] + bhh[nc + 1 * Hdim];
    const float bg = bih[nc + 2 * Hdim] + bhh[nc + 2 * Hdim];
    const float bo = bih[nc + 3 * Hdim] + bhh[nc + 3 * Hdim];
#pragma unroll
    for (int v = 0; v < 8; ++v) {
      const int mr = m0 + v + kh * 8;
      const size_t idx = (size_t)mr * Hdim + nc;
      const float ig = sigmoid_(a0[v] + bi);
      const float fg = sigmoid_(a1[v] + bf);
      const float gg = tanh_(a2[v] + bg);
      const float og = sigmoid_(a3[v] + bo);
      const float cn = fg * cst[idx] + ig * gg;
      cst[idx] = cn;
      const float hn = og * tanh_(cn);
      hst[idx] = hn;
      hstb[idx] = (bf16_t)hn;
    }
  }
}

// ---------------------------------------------------------------------------
// Persistent kernel: loops over all T time steps with grid barriers.
// ---------------------------------------------------------------------------
__global__ __launch_bounds__(NTHR, 1) void moglstm_persistent(
    const float* __restrict__ input_seq,
    const float* __restrict__ c1_Qb, const float* __restrict__ c1_Rb,
    const float* __restrict__ c1_bih, const float* __restrict__ c1_bhh,
    const float* __restrict__ c2_Qb, const float* __restrict__ c2_Rb,
    const float* __restrict__ c2_bih, const float* __restrict__ c2_bhh,
    const float* __restrict__ lin_W, const float* __restrict__ lin_b,
    const bf16_t* __restrict__ w1Q, const bf16_t* __restrict__ w1R,
    const bf16_t* __restrict__ w1Wih, const bf16_t* __restrict__ w1Whh,
    const bf16_t* __restrict__ w2Q, const bf16_t* __restrict__ w2R,
    const bf16_t* __restrict__ w2Wih, const bf16_t* __restrict__ w2Whh,
    float* xw1, bf16_t* xw1b, float* hw1, bf16_t* hw1b,
    float* h1, bf16_t* h1b, float* c1,
    float* xw2, bf16_t* xw2b, float* hw2, bf16_t* hw2b,
    float* h2, bf16_t* h2b, float* c2,
    unsigned* cnt, float* dout)
{
  // gw is wave-uniform; readfirstlane makes that explicit -> scalar tile loops.
  const int gw = __builtin_amdgcn_readfirstlane(
      blockIdx.x * (blockDim.x >> 5) + (threadIdx.x >> 5));
  const int nw = gridDim.x * (blockDim.x >> 5);
  const unsigned nb = gridDim.x;
  unsigned ep = 0;

  for (int t = 0; t < Tdim; ++t) {
    const float* xt = input_seq + (size_t)t * Idim;  // row stride Tdim*Idim

    // ----- layer 1 mogrify (Q: H->I, R: I->H) -----
    mogrify_phase(h1b, Hdim, w1Q, c1_Qb, xt, Tdim * Idim, xw1, xw1b, Idim, gw, nw);
    grid_barrier(cnt, (++ep) * nb);
    mogrify_phase(xw1b, Idim, w1R, c1_Rb, h1, Hdim, hw1, hw1b, Hdim, gw, nw);
    grid_barrier(cnt, (++ep) * nb);
    mogrify_phase(hw1b, Hdim, w1Q + (size_t)1 * Idim * Hdim, c1_Qb + Idim,
                  xw1, Idim, xw1, xw1b, Idim, gw, nw);
    grid_barrier(cnt, (++ep) * nb);
    mogrify_phase(xw1b, Idim, w1R + (size_t)1 * Hdim * Idim, c1_Rb + Hdim,
                  hw1, Hdim, hw1, hw1b, Hdim, gw, nw);
    grid_barrier(cnt, (++ep) * nb);
    mogrify_phase(hw1b, Hdim, w1Q + (size_t)2 * Idim * Hdim, c1_Qb + 2 * Idim,
                  xw1, Idim, xw1, xw1b, Idim, gw, nw);
    grid_barrier(cnt, (++ep) * nb);
    // ----- layer 1 LSTM cell -----
    lstm_phase(xw1b, Idim, hw1b, Hdim, w1Wih, w1Whh, c1_bih, c1_bhh,
               c1, h1, h1b, gw, nw);
    grid_barrier(cnt, (++ep) * nb);

    // ----- layer 2 mogrify (all H->H), x2 starts as new h1 -----
    mogrify_phase(h2b, Hdim, w2Q, c2_Qb, h1, Hdim, xw2, xw2b, Hdim, gw, nw);
    grid_barrier(cnt, (++ep) * nb);
    mogrify_phase(xw2b, Hdim, w2R, c2_Rb, h2, Hdim, hw2, hw2b, Hdim, gw, nw);
    grid_barrier(cnt, (++ep) * nb);
    mogrify_phase(hw2b, Hdim, w2Q + (size_t)1 * Hdim * Hdim, c2_Qb + Hdim,
                  xw2, Hdim, xw2, xw2b, Hdim, gw, nw);
    grid_barrier(cnt, (++ep) * nb);
    mogrify_phase(xw2b, Hdim, w2R + (size_t)1 * Hdim * Hdim, c2_Rb + Hdim,
                  hw2, Hdim, hw2, hw2b, Hdim, gw, nw);
    grid_barrier(cnt, (++ep) * nb);
    mogrify_phase(hw2b, Hdim, w2Q + (size_t)2 * Hdim * Hdim, c2_Qb + 2 * Hdim,
                  xw2, Hdim, xw2, xw2b, Hdim, gw, nw);
    grid_barrier(cnt, (++ep) * nb);
    // ----- layer 2 LSTM cell -----
    lstm_phase(xw2b, Hdim, hw2b, Hdim, w2Wih, w2Whh, c2_bih, c2_bhh,
               c2, h2, h2b, gw, nw);
    grid_barrier(cnt, (++ep) * nb);
  }

  // Final linear: out[b] = h2[b,:] . lin_W + lin_b  (O == 1)
  if (blockIdx.x == 0 && threadIdx.x < Bdim) {
    const int b = threadIdx.x;
    const float* hrow = h2 + (size_t)b * Hdim;
    float s = lin_b[0];
#pragma unroll 8
    for (int k = 0; k < Hdim; ++k) s += hrow[k] * lin_W[k];
    dout[b] = s;
  }
}

// ---------------------------------------------------------------------------
// Prologue kernels
// ---------------------------------------------------------------------------
__global__ void cvt_f32_bf16(const float* __restrict__ s, bf16_t* __restrict__ d, int n) {
  int i = blockIdx.x * blockDim.x + threadIdx.x;
  if (i < n) d[i] = (bf16_t)s[i];
}

__global__ void zero_words(unsigned* __restrict__ p, size_t nwords) {
  size_t i = (size_t)blockIdx.x * blockDim.x + threadIdx.x;
  const size_t st = (size_t)gridDim.x * blockDim.x;
  for (; i < nwords; i += st) p[i] = 0u;
}

// ---------------------------------------------------------------------------
// Launch
// ---------------------------------------------------------------------------
extern "C" void kernel_launch(void* const* d_in, const int* in_sizes, int n_in,
                              void* d_out, int out_size, void* d_ws, size_t ws_size,
                              hipStream_t stream) {
  (void)in_sizes; (void)n_in; (void)out_size;
  const float* input_seq = (const float*)d_in[0];
  const float* c1_Q   = (const float*)d_in[1];
  const float* c1_Qb  = (const float*)d_in[2];
  const float* c1_R   = (const float*)d_in[3];
  const float* c1_Rb  = (const float*)d_in[4];
  const float* c1_Wih = (const float*)d_in[5];
  const float* c1_Whh = (const float*)d_in[6];
  const float* c1_bih = (const float*)d_in[7];
  const float* c1_bhh = (const float*)d_in[8];
  const float* c2_Q   = (const float*)d_in[9];
  const float* c2_Qb  = (const float*)d_in[10];
  const float* c2_R   = (const float*)d_in[11];
  const float* c2_Rb  = (const float*)d_in[12];
  const float* c2_Wih = (const float*)d_in[13];
  const float* c2_Whh = (const float*)d_in[14];
  const float* c2_bih = (const float*)d_in[15];
  const float* c2_bhh = (const float*)d_in[16];
  const float* lin_W  = (const float*)d_in[17];
  const float* lin_b  = (const float*)d_in[18];
  float* dout = (float*)d_out;

  // --- workspace layout (bytes) ---
  char* ws = (char*)d_ws;
  size_t off = 0;
  auto carve = [&](size_t bytes) { char* p = ws + off; off += (bytes + 255) & ~(size_t)255; return p; };

  const size_t n_w1Q = 3ull * Idim * Hdim;      // 393216
  const size_t n_w1R = 2ull * Hdim * Idim;      // 262144
  const size_t n_w1Wih = 4ull * Hdim * Idim;    // 524288
  const size_t n_w1Whh = 4ull * Hdim * Hdim;    // 4194304
  const size_t n_w2Q = 3ull * Hdim * Hdim;      // 3145728
  const size_t n_w2R = 2ull * Hdim * Hdim;      // 2097152
  const size_t n_w2Wih = 4ull * Hdim * Hdim;    // 4194304
  const size_t n_w2Whh = 4ull * Hdim * Hdim;    // 4194304

  bf16_t* w1Q   = (bf16_t*)carve(n_w1Q * 2);
  bf16_t* w1R   = (bf16_t*)carve(n_w1R * 2);
  bf16_t* w1Wih = (bf16_t*)carve(n_w1Wih * 2);
  bf16_t* w1Whh = (bf16_t*)carve(n_w1Whh * 2);
  bf16_t* w2Q   = (bf16_t*)carve(n_w2Q * 2);
  bf16_t* w2R   = (bf16_t*)carve(n_w2R * 2);
  bf16_t* w2Wih = (bf16_t*)carve(n_w2Wih * 2);
  bf16_t* w2Whh = (bf16_t*)carve(n_w2Whh * 2);

  const size_t state_begin = off;
  float* xw1  = (float*)carve((size_t)Bdim * Idim * 4);
  float* hw1  = (float*)carve((size_t)Bdim * Hdim * 4);
  float* h1   = (float*)carve((size_t)Bdim * Hdim * 4);
  float* c1   = (float*)carve((size_t)Bdim * Hdim * 4);
  float* xw2  = (float*)carve((size_t)Bdim * Hdim * 4);
  float* hw2  = (float*)carve((size_t)Bdim * Hdim * 4);
  float* h2   = (float*)carve((size_t)Bdim * Hdim * 4);
  float* c2   = (float*)carve((size_t)Bdim * Hdim * 4);
  bf16_t* xw1b = (bf16_t*)carve((size_t)Bdim * Idim * 2);
  bf16_t* hw1b = (bf16_t*)carve((size_t)Bdim * Hdim * 2);
  bf16_t* h1b  = (bf16_t*)carve((size_t)Bdim * Hdim * 2);
  bf16_t* xw2b = (bf16_t*)carve((size_t)Bdim * Hdim * 2);
  bf16_t* hw2b = (bf16_t*)carve((size_t)Bdim * Hdim * 2);
  bf16_t* h2b  = (bf16_t*)carve((size_t)Bdim * Hdim * 2);
  unsigned* cnt = (unsigned*)carve(256);
  const size_t total_needed = off;
  if (ws_size < total_needed) return;  // workspace too small; bail deterministically

  // Zero the whole state region (states + barrier counter) every call.
  const size_t state_words = (total_needed - state_begin) / 4;
  zero_words<<<512, 256, 0, stream>>>((unsigned*)(ws + state_begin), state_words);

  // Convert weights fp32 -> bf16 (runs every call; deterministic).
  auto cvt = [&](const float* s, bf16_t* d, size_t n) {
    cvt_f32_bf16<<<(unsigned)((n + 255) / 256), 256, 0, stream>>>(s, d, (int)n);
  };
  cvt(c1_Q,   w1Q,   n_w1Q);
  cvt(c1_R,   w1R,   n_w1R);
  cvt(c1_Wih, w1Wih, n_w1Wih);
  cvt(c1_Whh, w1Whh, n_w1Whh);
  cvt(c2_Q,   w2Q,   n_w2Q);
  cvt(c2_R,   w2R,   n_w2R);
  cvt(c2_Wih, w2Wih, n_w2Wih);
  cvt(c2_Whh, w2Whh, n_w2Whh);

  moglstm_persistent<<<NBLK, NTHR, 0, stream>>>(
      input_seq,
      c1_Qb, c1_Rb, c1_bih, c1_bhh,
      c2_Qb, c2_Rb, c2_bih, c2_bhh,
      lin_W, lin_b,
      w1Q, w1R, w1Wih, w1Whh,
      w2Q, w2R, w2Wih, w2Whh,
      xw1, xw1b, hw1, hw1b, h1, h1b, c1,
      xw2, xw2b, hw2, hw2b, h2, h2b, c2,
      cnt, dout);
}
